// CausalSelfAttention_14998025797784
// MI455X (gfx1250) — matrix operand order repaired and checked by reading, not compiled, not run on hardware
//
#include <hip/hip_runtime.h>

typedef __attribute__((ext_vector_type(16))) __bf16 v16bf;
typedef __attribute__((ext_vector_type(8)))  __bf16 v8bf;
typedef __attribute__((ext_vector_type(8)))  float  v8f;

#define N_EMBD   1024
#define N_HEAD   16
#define HEAD_DIM 64
#define SEQ_T    2048
#define BATCH    4
#define M_TOT    (BATCH * SEQ_T)   // 8192
#define WPB      4                 // attention waves per block

// ---------------- fp32 -> bf16 (round to nearest even) ----------------
__device__ __forceinline__ __bf16 f2bf(float f) {
  union { float f; unsigned u; } uf; uf.f = f;
  unsigned u = uf.u;
  unsigned r = (u + 0x7FFFu + ((u >> 16) & 1u)) >> 16;
  union { unsigned short s; __bf16 b; } ub; ub.s = (unsigned short)r;
  return ub.b;
}

__global__ void cvt_f32_bf16(const float* __restrict__ src, __bf16* __restrict__ dst, int n) {
  int i = blockIdx.x * blockDim.x + threadIdx.x;
  int stride = gridDim.x * blockDim.x;
  for (; i < n; i += stride) dst[i] = f2bf(src[i]);
}

// ---------------- WMMA fragment helpers (ISA 7.12.2 layouts) ----------------
// A-matrix 16x32 bf16: lane<16 -> row=lane, K 0..7 & 16..23 ; lane>=16 -> row=lane-16, K 8..15 & 24..31
__device__ __forceinline__ v16bf load_a_frag(const __bf16* base, int lda, int lane) {
  int r  = lane & 15;
  int kb = (lane >> 4) << 3;          // 0 or 8
  const __bf16* p = base + (size_t)r * lda + kb;
  v8bf lo = *(const v8bf*)p;          // K kb..kb+7
  v8bf hi = *(const v8bf*)(p + 16);   // K kb+16..kb+23
  return __builtin_shufflevector(lo, hi, 0,1,2,3,4,5,6,7,8,9,10,11,12,13,14,15);
}

// B-matrix 32x16 bf16 stored column-major (i.e. row-major [N][K]):
// lane<16 -> col=lane, K 0..7 & 16..23 ; lane>=16 -> col=lane-16, K 8..15 & 24..31
__device__ __forceinline__ v16bf load_b_frag(const __bf16* base, int ldb, int lane) {
  int c  = lane & 15;
  int kb = (lane >> 4) << 3;          // 0 or 8
  const __bf16* p = base + (size_t)c * ldb + kb;
  v8bf lo = *(const v8bf*)p;
  v8bf hi = *(const v8bf*)(p + 16);
  return __builtin_shufflevector(lo, hi, 0,1,2,3,4,5,6,7,8,9,10,11,12,13,14,15);
}

__device__ __forceinline__ v8f wmma_bf16(v16bf a, v16bf b, v8f c) {
  return __builtin_amdgcn_wmma_f32_16x16x32_bf16(false, a, false, b, (short)0, c, false, false);
}

// ---------------- QKV GEMM: [8192,1024] x [3072,1024]^T, route to Q/K/V^T ----------------
__global__ void __launch_bounds__(32) qkv_gemm(const __bf16* __restrict__ X,
                                               const __bf16* __restrict__ W,
                                               const float*  __restrict__ bias,
                                               __bf16* __restrict__ Qb,
                                               __bf16* __restrict__ Kb,
                                               __bf16* __restrict__ Vt) {
  const int lane = threadIdx.x & 31;
  const int m0 = blockIdx.x * 64;
  const int n0 = blockIdx.y * 64;
  v8f acc[4][4] = {};
  for (int k0 = 0; k0 < N_EMBD; k0 += 32) {
    if (k0 + 32 < N_EMBD) {
      __builtin_prefetch(X + (size_t)m0 * N_EMBD + k0 + 32, 0, 1);
      __builtin_prefetch(W + (size_t)n0 * N_EMBD + k0 + 32, 0, 1);
    }
    v16bf a[4], b[4];
#pragma unroll
    for (int i = 0; i < 4; ++i) a[i] = load_a_frag(X + (size_t)(m0 + 16*i) * N_EMBD + k0, N_EMBD, lane);
#pragma unroll
    for (int j = 0; j < 4; ++j) b[j] = load_b_frag(W + (size_t)(n0 + 16*j) * N_EMBD + k0, N_EMBD, lane);
#pragma unroll
    for (int i = 0; i < 4; ++i)
#pragma unroll
      for (int j = 0; j < 4; ++j)
        acc[i][j] = wmma_bf16(a[i], b[j], acc[i][j]);
  }
  const int rbase = (lane >> 4) << 3;
  const int nlane = lane & 15;
#pragma unroll
  for (int i = 0; i < 4; ++i) {
#pragma unroll
    for (int j = 0; j < 4; ++j) {
      int n = n0 + 16*j + nlane;
      float bv = bias[n];
      int part = n >> 10;          // 0=q 1=k 2=v
      int ch = n & 1023;
      int h = ch >> 6, d = ch & 63;
#pragma unroll
      for (int e = 0; e < 8; ++e) {
        int m = m0 + 16*i + rbase + e;
        int bb = m >> 11, t = m & 2047;
        __bf16 val = f2bf(acc[i][j][e] + bv);
        size_t bh = (size_t)bb * N_HEAD + h;
        if (part == 0)      Qb[(bh * SEQ_T + t) * HEAD_DIM + d] = val;
        else if (part == 1) Kb[(bh * SEQ_T + t) * HEAD_DIM + d] = val;
        else                Vt[(bh * HEAD_DIM + d) * SEQ_T + t] = val;
      }
    }
  }
}

// ---------------- Flash attention: one 16-row query tile per wave ----------------
__global__ void __launch_bounds__(32 * WPB) attn_kernel(const __bf16* __restrict__ Qb,
                                                        const __bf16* __restrict__ Kb,
                                                        const __bf16* __restrict__ Vt,
                                                        __bf16* __restrict__ Y) {
  __shared__ __bf16 Psh[WPB][16 * 32];
  const int lane = threadIdx.x & 31;
  const int wave = threadIdx.x >> 5;
  const int bh = blockIdx.y;                       // 0..63
  const int t0 = (blockIdx.x * WPB + wave) * 16;
  const __bf16* q = Qb + (size_t)bh * SEQ_T * HEAD_DIM;
  const __bf16* k = Kb + (size_t)bh * SEQ_T * HEAD_DIM;
  const __bf16* v = Vt + (size_t)bh * HEAD_DIM * SEQ_T;   // [D][T]
  __bf16* Psl = &Psh[wave][0];

  v16bf aq0 = load_a_frag(q + (size_t)t0 * HEAD_DIM,      HEAD_DIM, lane);
  v16bf aq1 = load_a_frag(q + (size_t)t0 * HEAD_DIM + 32, HEAD_DIM, lane);

  v8f yacc[4] = {};
  float mrow[8], lrow[8];
#pragma unroll
  for (int e = 0; e < 8; ++e) { mrow[e] = -1e30f; lrow[e] = 0.f; }

  const float scale = 0.125f;                      // 1/sqrt(64)
  const int rbase = (lane >> 4) << 3;
  const int nlane = lane & 15;
  const int send = (t0 + 16 + 31) & ~31;           // diagonal included, 32-aligned

  for (int s0 = 0; s0 < send; s0 += 32) {
    // S = Q K^T for keys [s0, s0+32)
    v8f s0v = {}, s1v = {};
    v16bf bk;
    bk = load_b_frag(k + (size_t)s0 * HEAD_DIM,              HEAD_DIM, lane);
    s0v = wmma_bf16(aq0, bk, s0v);
    bk = load_b_frag(k + (size_t)s0 * HEAD_DIM + 32,         HEAD_DIM, lane);
    s0v = wmma_bf16(aq1, bk, s0v);
    bk = load_b_frag(k + (size_t)(s0 + 16) * HEAD_DIM,       HEAD_DIM, lane);
    s1v = wmma_bf16(aq0, bk, s1v);
    bk = load_b_frag(k + (size_t)(s0 + 16) * HEAD_DIM + 32,  HEAD_DIM, lane);
    s1v = wmma_bf16(aq1, bk, s1v);

    // online softmax; each C-tile row lives in one 16-lane half-group
#pragma unroll
    for (int e = 0; e < 8; ++e) {
      int t  = t0 + rbase + e;
      int sa = s0 + nlane;
      float x0 = (sa      <= t) ? s0v[e] * scale : -1e30f;
      float x1 = (sa + 16 <= t) ? s1v[e] * scale : -1e30f;
      float mx = fmaxf(x0, x1);
#pragma unroll
      for (int off = 1; off < 16; off <<= 1) mx = fmaxf(mx, __shfl_xor(mx, off, 32));
      float mnew = fmaxf(mrow[e], mx);
      float corr = __expf(mrow[e] - mnew);
      float p0 = __expf(x0 - mnew);
      float p1 = __expf(x1 - mnew);
      float rs = p0 + p1;
#pragma unroll
      for (int off = 1; off < 16; off <<= 1) rs += __shfl_xor(rs, off, 32);
      lrow[e] = lrow[e] * corr + rs;
      mrow[e] = mnew;
#pragma unroll
      for (int j = 0; j < 4; ++j) yacc[j][e] *= corr;
      int prow = rbase + e;
      Psl[prow * 32 + nlane]      = f2bf(p0);
      Psl[prow * 32 + 16 + nlane] = f2bf(p1);
    }
    // re-layout P (C-tile) -> A-fragment via LDS (wave-private slab, DS ops in-order)
    v16bf pf = load_a_frag(Psl, 32, lane);
#pragma unroll
    for (int j = 0; j < 4; ++j) {
      v16bf bv = load_b_frag(v + (size_t)(16 * j) * SEQ_T + s0, SEQ_T, lane);
      yacc[j] = wmma_bf16(pf, bv, yacc[j]);
    }
  }

  const int b = bh >> 4, h = bh & 15;
#pragma unroll
  for (int e = 0; e < 8; ++e) {
    int t = t0 + rbase + e;
    size_t row = ((size_t)b * SEQ_T + t) * N_EMBD + h * HEAD_DIM;
    float inv = 1.0f / lrow[e];
#pragma unroll
    for (int j = 0; j < 4; ++j)
      Y[row + 16 * j + nlane] = f2bf(yacc[j][e] * inv);
  }
}

// ---------------- output projection: [8192,1024] x [1024,1024]^T + bias -> fp32 ----------------
__global__ void __launch_bounds__(32) proj_gemm(const __bf16* __restrict__ Yb,
                                                const __bf16* __restrict__ W,
                                                const float*  __restrict__ bias,
                                                float* __restrict__ Out) {
  const int lane = threadIdx.x & 31;
  const int m0 = blockIdx.x * 64;
  const int n0 = blockIdx.y * 64;
  v8f acc[4][4] = {};
  for (int k0 = 0; k0 < N_EMBD; k0 += 32) {
    if (k0 + 32 < N_EMBD) {
      __builtin_prefetch(Yb + (size_t)m0 * N_EMBD + k0 + 32, 0, 1);
      __builtin_prefetch(W  + (size_t)n0 * N_EMBD + k0 + 32, 0, 1);
    }
    v16bf a[4], b[4];
#pragma unroll
    for (int i = 0; i < 4; ++i) a[i] = load_a_frag(Yb + (size_t)(m0 + 16*i) * N_EMBD + k0, N_EMBD, lane);
#pragma unroll
    for (int j = 0; j < 4; ++j) b[j] = load_b_frag(W + (size_t)(n0 + 16*j) * N_EMBD + k0, N_EMBD, lane);
#pragma unroll
    for (int i = 0; i < 4; ++i)
#pragma unroll
      for (int j = 0; j < 4; ++j)
        acc[i][j] = wmma_bf16(a[i], b[j], acc[i][j]);
  }
  const int rbase = (lane >> 4) << 3;
  const int nlane = lane & 15;
#pragma unroll
  for (int i = 0; i < 4; ++i)
#pragma unroll
    for (int j = 0; j < 4; ++j) {
      int n = n0 + 16*j + nlane;
      float bv = bias[n];
#pragma unroll
      for (int e = 0; e < 8; ++e) {
        int m = m0 + 16*i + rbase + e;
        Out[(size_t)m * N_EMBD + n] = acc[i][j][e] + bv;
      }
    }
}

// ---------------- launcher ----------------
extern "C" void kernel_launch(void* const* d_in, const int* in_sizes, int n_in,
                              void* d_out, int out_size, void* d_ws, size_t ws_size,
                              hipStream_t stream) {
  const float* x      = (const float*)d_in[0];
  const float* W_attn = (const float*)d_in[1];
  const float* b_attn = (const float*)d_in[2];
  const float* W_proj = (const float*)d_in[3];
  const float* b_proj = (const float*)d_in[4];
  float* out = (float*)d_out;

  char* ws = (char*)d_ws;
  size_t off = 0;
  auto alloc = [&](size_t bytes) -> void* {
    void* p = ws + off;
    off += (bytes + 255) & ~(size_t)255;
    return p;
  };
  __bf16* xb  = (__bf16*)alloc((size_t)M_TOT * N_EMBD * 2);       // 16 MB
  __bf16* wab = (__bf16*)alloc((size_t)3 * N_EMBD * N_EMBD * 2);  //  6 MB
  __bf16* wpb = (__bf16*)alloc((size_t)N_EMBD * N_EMBD * 2);      //  2 MB
  __bf16* qb  = (__bf16*)alloc((size_t)M_TOT * N_EMBD * 2);       // 16 MB
  __bf16* kb  = (__bf16*)alloc((size_t)M_TOT * N_EMBD * 2);       // 16 MB
  __bf16* vt  = (__bf16*)alloc((size_t)M_TOT * N_EMBD * 2);       // 16 MB
  __bf16* yb  = (__bf16*)alloc((size_t)M_TOT * N_EMBD * 2);       // 16 MB

  cvt_f32_bf16<<<dim3(2048), 256, 0, stream>>>(x,      xb,  M_TOT * N_EMBD);
  cvt_f32_bf16<<<dim3(1024), 256, 0, stream>>>(W_attn, wab, 3 * N_EMBD * N_EMBD);
  cvt_f32_bf16<<<dim3(512),  256, 0, stream>>>(W_proj, wpb, N_EMBD * N_EMBD);

  qkv_gemm<<<dim3(M_TOT / 64, (3 * N_EMBD) / 64), 32, 0, stream>>>(xb, wab, b_attn, qb, kb, vt);

  attn_kernel<<<dim3(SEQ_T / (16 * WPB), BATCH * N_HEAD), 32 * WPB, 0, stream>>>(qb, kb, vt, yb);

  proj_gemm<<<dim3(M_TOT / 64, N_EMBD / 64), 32, 0, stream>>>(yb, wpb, b_proj, out);
}
